// MoELayer_89799176225238
// MI455X (gfx1250) — matrix operand order repaired and checked by reading, not compiled, not run on hardware
//
#include <hip/hip_runtime.h>
#include <hip/hip_bf16.h>
#include <math.h>

// ---------------------------------------------------------------------------
// MoE (top-2 of 8 experts), D=1024, F=4096, T=8192 tokens.
// bf16 WMMA (v_wmma_f32_16x16x32_bf16), async global->LDS staging (CDNA5),
// double-buffered LDS, fragment-ready swizzled bf16 weights.
// ---------------------------------------------------------------------------

#define T_TOK   8192
#define DDIM    1024
#define FDIM    4096
#define NEXP    8
#define OUT_N   (T_TOK * DDIM)        // loss scalar lives at index OUT_N

typedef __bf16 v16bf __attribute__((ext_vector_type(16)));
typedef float  v8f   __attribute__((ext_vector_type(8)));

union ABFrag  { v16bf v; unsigned u[8]; };
union AccFrag { v8f v; float f[8]; };

__device__ __forceinline__ unsigned short f2bf(float f) {
    unsigned u = __float_as_uint(f);
    unsigned rnd = 0x7FFFu + ((u >> 16) & 1u);    // round-to-nearest-even
    return (unsigned short)((u + rnd) >> 16);
}
__device__ __forceinline__ unsigned pack2bf(float lo, float hi) {
    return (unsigned)f2bf(lo) | ((unsigned)f2bf(hi) << 16);
}

// LDS byte offset of a __shared__ pointer (generic addr low 32 bits)
__device__ __forceinline__ unsigned lds_off(const void* p) {
    return (unsigned)(uintptr_t)p;
}
// CDNA5 async DMA: 16B global -> LDS, tracked by ASYNCcnt (no VGPR data)
__device__ __forceinline__ void async_b128(unsigned lds, const void* g) {
    asm volatile("global_load_async_to_lds_b128 %0, %1, off"
                 :: "v"(lds), "v"((unsigned long long)(uintptr_t)g)
                 : "memory");
}

// ---------------------------------------------------------------------------
// 0) zero output
// ---------------------------------------------------------------------------
__global__ void zero_kernel(float* __restrict__ p, int n) {
    int i = blockIdx.x * blockDim.x + threadIdx.x;
    if (i < n) p[i] = 0.0f;
}

// ---------------------------------------------------------------------------
// bf16 pre-pack kernels (once per launch, weights stay L2-resident)
// ---------------------------------------------------------------------------
// X: elementwise f32 -> bf16 (pairs along contiguous k)
__global__ void pack_x_kernel(const float* __restrict__ src,
                              unsigned* __restrict__ dst, int npairs) {
    int i = blockIdx.x * blockDim.x + threadIdx.x;
    if (i < npairs) dst[i] = pack2bf(src[2 * i], src[2 * i + 1]);
}
// Weights -> fragment-ready swizzle: dst[e][kt][S][lh][lm][j] =
//   pair{ w[e][kt*32+lh*16+2j][S*16+lm], w[e][...+1][S*16+lm] }
// K, N are the per-expert GEMM dims (w1: K=1024,N=4096; w2: K=4096,N=1024).
__global__ void pack_w_swz_kernel(const float* __restrict__ src,
                                  unsigned* __restrict__ dst,
                                  int K, int N, int total) {
    int i = blockIdx.x * blockDim.x + threadIdx.x;
    if (i >= total) return;
    int j  = i & 7;
    int lm = (i >> 3) & 15;
    int lh = (i >> 7) & 1;
    int S  = (i >> 8) % (N / 16);
    int rest = (i >> 8) / (N / 16);
    int kt = rest % (K / 32);
    int e  = rest / (K / 32);
    int n  = S * 16 + lm;
    int k  = kt * 32 + lh * 16 + 2 * j;
    const float* s0 = src + ((size_t)e * K + k) * N + n;
    dst[i] = pack2bf(s0[0], s0[N]);
}

// ---------------------------------------------------------------------------
// 1) router: one wave32 per token
// ---------------------------------------------------------------------------
__global__ void router_kernel(const float* __restrict__ hs,
                              const float* __restrict__ rw,
                              int*   __restrict__ topk_idx,
                              float* __restrict__ topk_w,
                              float* __restrict__ partials) {
    const int wave = threadIdx.x >> 5, lane = threadIdx.x & 31;
    const int token = blockIdx.x * 8 + wave;

    float acc[NEXP];
#pragma unroll
    for (int e = 0; e < NEXP; ++e) acc[e] = 0.0f;
    const float* x = hs + (size_t)token * DDIM;
    for (int k = lane; k < DDIM; k += 32) {
        float xv = x[k];
        const float* r = rw + k * NEXP;
#pragma unroll
        for (int e = 0; e < NEXP; ++e) acc[e] = fmaf(xv, r[e], acc[e]);
    }
#pragma unroll
    for (int e = 0; e < NEXP; ++e)
        for (int o = 16; o; o >>= 1) acc[e] += __shfl_xor(acc[e], o, 32);

    __shared__ float sprob[8][NEXP];
    if (lane == 0) {
        float mx = acc[0];
#pragma unroll
        for (int e = 1; e < NEXP; ++e) mx = fmaxf(mx, acc[e]);
        float p[NEXP], s = 0.0f;
#pragma unroll
        for (int e = 0; e < NEXP; ++e) { p[e] = expf(acc[e] - mx); s += p[e]; }
        float inv = 1.0f / s;
#pragma unroll
        for (int e = 0; e < NEXP; ++e) { p[e] *= inv; sprob[wave][e] = p[e]; }
        int a0 = 0;
#pragma unroll
        for (int e = 1; e < NEXP; ++e) if (p[e] > p[a0]) a0 = e;
        int a1 = (a0 == 0) ? 1 : 0;
#pragma unroll
        for (int e = 0; e < NEXP; ++e) if (e != a0 && p[e] > p[a1]) a1 = e;
        float s2 = p[a0] + p[a1];
        topk_idx[2 * token] = a0;  topk_idx[2 * token + 1] = a1;
        topk_w[2 * token] = p[a0] / s2;  topk_w[2 * token + 1] = p[a1] / s2;
    }
    __syncthreads();
    if (threadIdx.x < NEXP) {                    // fixed-order -> deterministic
        float s = 0.0f;
#pragma unroll
        for (int w = 0; w < 8; ++w) s += sprob[w][threadIdx.x];
        partials[blockIdx.x * NEXP + threadIdx.x] = s;
    }
}

// ---------------------------------------------------------------------------
// 2) load-balancing loss
// ---------------------------------------------------------------------------
__global__ void lb_kernel(const float* __restrict__ partials, int nblk,
                          float* __restrict__ out_loss) {
    __shared__ float load[NEXP];
    if (threadIdx.x < NEXP) {
        float s = 0.0f;
        for (int b = 0; b < nblk; ++b) s += partials[b * NEXP + threadIdx.x];
        load[threadIdx.x] = s / (float)T_TOK;
    }
    __syncthreads();
    if (threadIdx.x == 0) {
        float a = 0.0f;
#pragma unroll
        for (int e = 0; e < NEXP; ++e) {
            float d = load[e] - 1.0f / (float)NEXP;  a += d * d;
        }
        out_loss[0] = (a / (float)NEXP) * 0.001f;
    }
}

// ---------------------------------------------------------------------------
// 3) compaction: per-expert slot lists, ascending token order (deterministic)
// ---------------------------------------------------------------------------
__global__ void compact_kernel(const int* __restrict__ tidx,
                               const float* __restrict__ tw,
                               int* __restrict__ offs,
                               int* __restrict__ slot_token,
                               float* __restrict__ slot_w) {
    __shared__ int counts[NEXP];
    __shared__ int soffs[NEXP + 1];
    const int wave = threadIdx.x >> 5, lane = threadIdx.x & 31;

    int c = 0;
    for (int t = lane; t < T_TOK; t += 32)
        c += (tidx[2 * t] == wave) + (tidx[2 * t + 1] == wave);
    for (int o = 16; o; o >>= 1) c += __shfl_xor(c, o, 32);
    if (lane == 0) counts[wave] = c;
    __syncthreads();
    if (threadIdx.x == 0) {
        soffs[0] = 0;
        for (int e = 0; e < NEXP; ++e) soffs[e + 1] = soffs[e] + counts[e];
    }
    __syncthreads();

    int base = soffs[wave];
    for (int chunk = 0; chunk < T_TOK; chunk += 32) {
        int t = chunk + lane;
        int e0 = tidx[2 * t], e1 = tidx[2 * t + 1];
        bool m = (e0 == wave) || (e1 == wave);
        unsigned mask = (unsigned)__ballot(m);
        if (m) {
            int pos = __popc(mask & ((1u << lane) - 1u));
            slot_token[base + pos] = t;
            slot_w[base + pos] = (e0 == wave) ? tw[2 * t] : tw[2 * t + 1];
        }
        base += __popc(mask);
    }
    if (threadIdx.x <= NEXP) offs[threadIdx.x] = soffs[threadIdx.x];
}

// ---------------------------------------------------------------------------
// WMMA tiling: M_TILE=256 (8 waves x 2 m-subtiles) x N_TILE=128, K-step 32.
// A in LDS: bf16 rows [256][ASTR], 80B stride, 2x ds_load_b128 per fragment.
// B in LDS: fragment-ready flat 2048 u32 (8KB), 2x ds_load_b128 per fragment,
//           filled by one contiguous global chunk (async b128 copies).
// ---------------------------------------------------------------------------
#define ASTR  40      // u16 per A row (80B, multiple of 16B)
#define MTILE 256
#define MAX_MT 32     // 8192/256
#define BTILE_U32 2048  // 16 kp x 128 n pair-dwords per B tile

// A fragment 16x32 bf16: u[j] = K pair {kb, kb+1}, kb=2*j+16*lh
__device__ __forceinline__ void load_a_frag(ABFrag& a,
                                            const unsigned short* row, int lh) {
    uint4 x0 = *(const uint4*)(row + lh * 16);       // k = 16lh .. 16lh+7
    uint4 x1 = *(const uint4*)(row + lh * 16 + 8);   // k = 16lh+8 .. 16lh+15
    a.u[0] = x0.x; a.u[1] = x0.y; a.u[2] = x0.z; a.u[3] = x0.w;
    a.u[4] = x1.x; a.u[5] = x1.y; a.u[6] = x1.z; a.u[7] = x1.w;
}
// B fragment 32x16 bf16 (fragment-ready layout): lane's 8 dwords contiguous
__device__ __forceinline__ void load_b_frag(ABFrag& b, const unsigned* Bb,
                                            int sub, int lane) {
    const uint4* p = (const uint4*)(Bb + sub * 256 + lane * 8);
    uint4 x0 = p[0], x1 = p[1];
    b.u[0] = x0.x; b.u[1] = x0.y; b.u[2] = x0.z; b.u[3] = x0.w;
    b.u[4] = x1.x; b.u[5] = x1.y; b.u[6] = x1.z; b.u[7] = x1.w;
}

#define WMMA_BF16(A, B, C) \
    __builtin_amdgcn_wmma_f32_16x16x32_bf16(false, (A), false, (B), (short)0, (C), false, false)

// shared main loop body (async double-buffered). KT = #k-steps.
#define GEMM_MAINLOOP(KT)                                                     \
    {                                                                         \
        issue(0, 0);                                                          \
        for (int kt = 0; kt < (KT); ++kt) {                                   \
            int buf = kt & 1;                                                 \
            if (kt + 1 < (KT)) {                                              \
                issue(kt + 1, buf ^ 1);                                       \
                asm volatile("s_wait_asynccnt 0x6" ::: "memory");             \
            } else {                                                          \
                asm volatile("s_wait_asynccnt 0x0" ::: "memory");             \
            }                                                                 \
            __syncthreads();                                                  \
            const unsigned short* Ab = &As[buf][0];                           \
            const unsigned* Bb = &Bs[buf][0];                                 \
            ABFrag afr0, afr1;                                                \
            load_a_frag(afr0, Ab + (wave * 32 + lm) * ASTR, lh);              \
            load_a_frag(afr1, Ab + (wave * 32 + 16 + lm) * ASTR, lh);         \
            _Pragma("unroll")                                                 \
            for (int s = 0; s < 8; ++s) {                                     \
                ABFrag bfr;                                                   \
                load_b_frag(bfr, Bb, s, lane);                                \
                acc[0][s].v = WMMA_BF16(afr0.v, bfr.v, acc[0][s].v);          \
                acc[1][s].v = WMMA_BF16(afr1.v, bfr.v, acc[1][s].v);          \
            }                                                                 \
            __syncthreads();                                                  \
        }                                                                     \
    }

// ---------------------------------------------------------------------------
// 4) gemm1: H = gelu(gather(Xbf16) @ w1[e] + b1[e]) -> bf16 H
//    grid.x = E*MAX_MT, grid.y = F/128
// ---------------------------------------------------------------------------
__global__ void __launch_bounds__(256)
gemm1_kernel(const unsigned short* __restrict__ xp,   // [T][D] bf16
             const unsigned* __restrict__ w1s,        // swizzled pair-dwords
             const float* __restrict__ b1,
             const int* __restrict__ offs,
             const int* __restrict__ slot_token,
             unsigned short* __restrict__ H) {
    const int e   = blockIdx.x >> 5;
    const int mt  = blockIdx.x & (MAX_MT - 1);
    const int cnt = offs[e + 1] - offs[e];
    const int row0 = mt * MTILE;
    if (row0 >= cnt) return;                       // uniform per block
    const int rows = min(MTILE, cnt - row0);
    const int base_slot = offs[e] + row0;
    const int f0 = blockIdx.y * 128;

    __shared__ __align__(16) unsigned short As[2][MTILE * ASTR];
    __shared__ __align__(16) unsigned Bs[2][BTILE_U32];

    const int tid  = threadIdx.x;
    const int wave = tid >> 5, lane = tid & 31;
    const int lm = lane & 15, lh = lane >> 4;

    // A staging: invalid rows alias the last valid row; their outputs are
    // discarded by the row<rows guard, so no zero-fill is needed.
    const int arow = tid;
    const int tok = slot_token[base_slot + min(arow, rows - 1)];
    const unsigned short* a_src = xp + (size_t)tok * DDIM;
    const unsigned a_lds0 = lds_off(&As[0][arow * ASTR]);
    const unsigned a_lds1 = lds_off(&As[1][arow * ASTR]);

    // B staging: contiguous 8KB chunk per k-step; thread copies 32B.
    // chunk(kt) = w1s + ((e*KT + kt)*S_TOT + S0)*256 ; S_TOT=F/16, S0=f0/16
    const unsigned* b_base =
        w1s + ((size_t)e * (DDIM / 32) * (FDIM / 16) + (size_t)(f0 >> 4)) * 256 + tid * 8;
    const unsigned b_lds0 = lds_off(&Bs[0][tid * 8]);
    const unsigned b_lds1 = lds_off(&Bs[1][tid * 8]);

    AccFrag acc[2][8];
#pragma unroll
    for (int m = 0; m < 2; ++m)
#pragma unroll
        for (int s = 0; s < 8; ++s)
#pragma unroll
            for (int i = 0; i < 8; ++i) acc[m][s].f[i] = 0.0f;

    auto issue = [&](int kt, int buf) {
        const unsigned short* ga = a_src + kt * 32;
        unsigned la = buf ? a_lds1 : a_lds0;
#pragma unroll
        for (int q = 0; q < 4; ++q) async_b128(la + q * 16, ga + q * 8);
        const unsigned* gb = b_base + (size_t)kt * (FDIM / 16) * 256;
        unsigned lb = buf ? b_lds1 : b_lds0;
        async_b128(lb, gb);
        async_b128(lb + 16, gb + 4);
    };

    GEMM_MAINLOOP(DDIM / 32)

    // epilogue: bias + exact GELU -> bf16 H
#pragma unroll
    for (int m = 0; m < 2; ++m)
#pragma unroll
        for (int s = 0; s < 8; ++s) {
            int f = f0 + s * 16 + lm;
            float bias = b1[e * FDIM + f];
#pragma unroll
            for (int r = 0; r < 8; ++r) {
                int row = wave * 32 + m * 16 + lh * 8 + r;
                if (row < rows) {
                    float x = acc[m][s].f[r] + bias;
                    float gx = 0.5f * x * (1.0f + erff(x * 0.70710678118654752f));
                    H[(size_t)(base_slot + row) * FDIM + f] = f2bf(gx);
                }
            }
        }
}

// ---------------------------------------------------------------------------
// 5) gemm2: out += combine_w * (H @ w2[e] + b2[e])   (f32 atomic scatter-add,
//    exactly 2 contributions per element -> bitwise deterministic)
//    grid.x = E*MAX_MT, grid.y = D/128
// ---------------------------------------------------------------------------
__global__ void __launch_bounds__(256)
gemm2_kernel(const unsigned short* __restrict__ H,    // [2T][F] bf16
             const unsigned* __restrict__ w2s,        // swizzled pair-dwords
             const float* __restrict__ b2,
             const int* __restrict__ offs,
             const int* __restrict__ slot_token,
             const float* __restrict__ slot_w,
             float* __restrict__ out) {
    const int e   = blockIdx.x >> 5;
    const int mt  = blockIdx.x & (MAX_MT - 1);
    const int cnt = offs[e + 1] - offs[e];
    const int row0 = mt * MTILE;
    if (row0 >= cnt) return;
    const int rows = min(MTILE, cnt - row0);
    const int base_slot = offs[e] + row0;
    const int n0 = blockIdx.y * 128;

    __shared__ __align__(16) unsigned short As[2][MTILE * ASTR];
    __shared__ __align__(16) unsigned Bs[2][BTILE_U32];

    const int tid  = threadIdx.x;
    const int wave = tid >> 5, lane = tid & 31;
    const int lm = lane & 15, lh = lane >> 4;

    const int arow = tid;
    const int aslot = base_slot + min(arow, rows - 1);
    const unsigned short* a_src = H + (size_t)aslot * FDIM;
    const unsigned a_lds0 = lds_off(&As[0][arow * ASTR]);
    const unsigned a_lds1 = lds_off(&As[1][arow * ASTR]);

    const unsigned* b_base =
        w2s + ((size_t)e * (FDIM / 32) * (DDIM / 16) + (size_t)(n0 >> 4)) * 256 + tid * 8;
    const unsigned b_lds0 = lds_off(&Bs[0][tid * 8]);
    const unsigned b_lds1 = lds_off(&Bs[1][tid * 8]);

    AccFrag acc[2][8];
#pragma unroll
    for (int m = 0; m < 2; ++m)
#pragma unroll
        for (int s = 0; s < 8; ++s)
#pragma unroll
            for (int i = 0; i < 8; ++i) acc[m][s].f[i] = 0.0f;

    auto issue = [&](int kt, int buf) {
        const unsigned short* ga = a_src + kt * 32;
        unsigned la = buf ? a_lds1 : a_lds0;
#pragma unroll
        for (int q = 0; q < 4; ++q) async_b128(la + q * 16, ga + q * 8);
        const unsigned* gb = b_base + (size_t)kt * (DDIM / 16) * 256;
        unsigned lb = buf ? b_lds1 : b_lds0;
        async_b128(lb, gb);
        async_b128(lb + 16, gb + 4);
    };

    GEMM_MAINLOOP(FDIM / 32)

#pragma unroll
    for (int m = 0; m < 2; ++m)
#pragma unroll
        for (int s = 0; s < 8; ++s) {
            int n = n0 + s * 16 + lm;
            float bias = b2[e * DDIM + n];
#pragma unroll
            for (int r = 0; r < 8; ++r) {
                int row = wave * 32 + m * 16 + lh * 8 + r;
                if (row < rows) {
                    int slot = base_slot + row;
                    float val = slot_w[slot] * (acc[m][s].f[r] + bias);
                    atomicAdd(&out[(size_t)slot_token[slot] * DDIM + n], val);
                }
            }
        }
}

// ---------------------------------------------------------------------------
// launch
// ---------------------------------------------------------------------------
extern "C" void kernel_launch(void* const* d_in, const int* in_sizes, int n_in,
                              void* d_out, int out_size, void* d_ws, size_t ws_size,
                              hipStream_t stream) {
    const float* hs = (const float*)d_in[0];
    const float* rw = (const float*)d_in[1];
    const float* w1 = (const float*)d_in[2];
    const float* b1 = (const float*)d_in[3];
    const float* w2 = (const float*)d_in[4];
    const float* b2 = (const float*)d_in[5];
    float* out = (float*)d_out;

    char* ws = (char*)d_ws;
    size_t off = 0;
    auto take = [&](size_t bytes) {
        void* p = ws + off; off += (bytes + 255) & ~(size_t)255; return p;
    };
    int*            topk_idx   = (int*)           take((size_t)2 * T_TOK * 4);
    float*          topk_w     = (float*)         take((size_t)2 * T_TOK * 4);
    float*          partials   = (float*)         take((size_t)1024 * NEXP * 4);
    int*            offs       = (int*)           take((size_t)16 * 4);
    int*            slot_token = (int*)           take((size_t)2 * T_TOK * 4);
    float*          slot_w     = (float*)         take((size_t)2 * T_TOK * 4);
    unsigned short* xp         = (unsigned short*)take((size_t)T_TOK * DDIM * 2);
    unsigned*       w1s        = (unsigned*)      take((size_t)NEXP * 512 * FDIM * 4);
    unsigned*       w2s        = (unsigned*)      take((size_t)NEXP * 2048 * DDIM * 4);
    unsigned short* H          = (unsigned short*)take((size_t)2 * T_TOK * FDIM * 2);
    (void)ws_size; (void)n_in; (void)in_sizes; (void)out_size;

    const int n_out = OUT_N + 1;
    zero_kernel<<<(n_out + 255) / 256, 256, 0, stream>>>(out, n_out);

    // bf16 pre-pack (fragment-ready weight swizzle)
    {
        int nx = T_TOK * DDIM / 2;
        pack_x_kernel<<<(nx + 255) / 256, 256, 0, stream>>>(hs, (unsigned*)xp, nx);
        int n1 = NEXP * (DDIM / 2) * FDIM;
        pack_w_swz_kernel<<<(n1 + 255) / 256, 256, 0, stream>>>(w1, w1s, DDIM, FDIM, n1);
        int n2 = NEXP * (FDIM / 2) * DDIM;
        pack_w_swz_kernel<<<(n2 + 255) / 256, 256, 0, stream>>>(w2, w2s, FDIM, DDIM, n2);
    }

    router_kernel<<<T_TOK / 8, 256, 0, stream>>>(hs, rw, topk_idx, topk_w, partials);
    lb_kernel<<<1, 32, 0, stream>>>(partials, T_TOK / 8, out + OUT_N);
    compact_kernel<<<1, 256, 0, stream>>>(topk_idx, topk_w, offs, slot_token, slot_w);

    gemm1_kernel<<<dim3(NEXP * MAX_MT, FDIM / 128), 256, 0, stream>>>(
        xp, w1s, b1, offs, slot_token, H);
    gemm2_kernel<<<dim3(NEXP * MAX_MT, DDIM / 128), 256, 0, stream>>>(
        H, w2s, b2, offs, slot_token, slot_w, out);
}
